// HMP_29188597744146
// MI455X (gfx1250) — matrix operand order repaired and checked by reading, not compiled, not run on hardware
//
#include <hip/hip_runtime.h>
#include <math.h>

#define UDIM 128
static const long long NN1 = 400000, NN2 = 400000, NN3 = 800000, NN4 = 1200000;

typedef __attribute__((ext_vector_type(16))) __bf16 v16bf;
typedef __attribute__((ext_vector_type(8)))  __bf16 v8bf;
typedef __attribute__((ext_vector_type(8)))  float  v8f;
typedef __attribute__((ext_vector_type(16))) float  v16f;

union FragAB { v16bf v; unsigned int w[8]; };
union U8bf   { v8bf v; uint4 q; unsigned int w[4]; };

__device__ __forceinline__ unsigned short f2bf(float f) {
    unsigned int u = __float_as_uint(f);
    u += 0x7fffu + ((u >> 16) & 1u);
    return (unsigned short)(u >> 16);
}
__device__ __forceinline__ float fast_tanh(float x) {
#if __has_builtin(__builtin_amdgcn_tanhf)
    return __builtin_amdgcn_tanhf(x);
#else
    return tanhf(x);
#endif
}

// ---------------------------------------------------------------------------
// Pack W (K x 128 row-major f32) into A-fragment-swizzled bf16.
// A-frag K map (16-bit A 16x32): half h: element e -> 16h+e (contiguous)
// ---------------------------------------------------------------------------
__global__ void pack_wT_kernel(const float* __restrict__ W,
                               unsigned short* __restrict__ out, int KC) {
    int t = blockIdx.x * 256 + threadIdx.x;
    if (t >= KC * 4096) return;
    int e    = t & 15;
    int lane = (t >> 4) & 31;
    int m    = (t >> 9) & 7;
    int kc   = t >> 12;
    int h    = lane >> 4;
    int K    = kc * 32 + 16 * h + e;
    int n    = m * 16 + (lane & 15);
    out[t] = f2bf(W[K * UDIM + n]);
}

__global__ void zero4_kernel(float4* __restrict__ p, long long n4) {
    long long t = (long long)blockIdx.x * 256 + threadIdx.x;
    if (t < n4) p[t] = float4{0.f, 0.f, 0.f, 0.f};
}

// scatter-add of a bf16 matrix into an f32 accumulator (both index columns)
__global__ void scatter_add_bf16_kernel(const unsigned int* __restrict__ src,
                                        const int* __restrict__ idx,
                                        float* __restrict__ dst,
                                        long long nsrc) {
    long long t = (long long)blockIdx.x * 256 + threadIdx.x;
    if (t >= nsrc * 64) return;
    long long row = t >> 6;
    int cp = (int)(t & 63);
    unsigned int u = src[t];                       // 2 bf16 = cols 2cp, 2cp+1
    float v0 = __uint_as_float(u << 16);
    float v1 = __uint_as_float(u & 0xffff0000u);
    const int2 ii = *(const int2*)(idx + 2 * row);
    float* d0 = dst + (long long)ii.x * UDIM + cp * 2;
    float* d1 = dst + (long long)ii.y * UDIM + cp * 2;
    unsafeAtomicAdd(d0,     v0);
    unsafeAtomicAdd(d0 + 1, v1);
    unsafeAtomicAdd(d1,     v0);
    unsafeAtomicAdd(d1 + 1, v1);
}

// B-fragment load: 16 consecutive elements of one row at element offset `off`
template <bool BF>
__device__ __forceinline__ void load_bfrag(FragAB& xb, const char* rowp, int off) {
    if constexpr (BF) {
        const uint4* q = (const uint4*)(rowp + off * 2);        // 32B raw
        const uint4 a = q[0], b = q[1];
        xb.w[0] = a.x; xb.w[1] = a.y; xb.w[2] = a.z; xb.w[3] = a.w;
        xb.w[4] = b.x; xb.w[5] = b.y; xb.w[6] = b.z; xb.w[7] = b.w;
    } else {
        const v16f xf = *(const v16f*)(rowp + off * 4);         // 64B load
        xb.v = __builtin_convertvector(xf, v16bf);              // fptrunc x16
    }
}

// weight A-fragment from LDS: two ds_load_b128 at immediate offsets
__device__ __forceinline__ void load_wfrag(FragAB& wa, const unsigned int* base,
                                           int idx) {
    const uint4* wp = (const uint4*)(base + idx);
    const uint4 lo = wp[0], hi = wp[1];
    wa.w[0] = lo.x; wa.w[1] = lo.y; wa.w[2] = lo.z; wa.w[3] = lo.w;
    wa.w[4] = hi.x; wa.w[5] = hi.y; wa.w[6] = hi.z; wa.w[7] = hi.w;
}

// ---------------------------------------------------------------------------
// Fused MLP stage:  out = tanh(X @ W1 + b1) @ W2 + b2   (transposed WMMA form)
// SBF/OBF/OUTBF select bf16 vs f32 for self-rows / other-rows / output.
// ---------------------------------------------------------------------------
template <int KC1, bool SBF, bool OBF, bool OUTBF>
__global__ __launch_bounds__(256) void mlp_stage_kernel(
    const void* __restrict__ selfp, const void* __restrict__ other,
    const int* __restrict__ idx, const float* __restrict__ b1,
    const float* __restrict__ b2, const unsigned int* __restrict__ w1p,
    const unsigned int* __restrict__ w2p, void* __restrict__ out) {
    extern __shared__ unsigned int lds[];
    const int tid  = threadIdx.x;
    const int lane = tid & 31;
    const int wave = tid >> 5;
    const int hf   = lane >> 4;
    const int ln   = lane & 15;
    constexpr int W1U = KC1 * 8 * 32 * 8;
    constexpr int W2U = 4 * 8 * 32 * 8;

    {   // cooperative contiguous copy of pre-swizzled weights into LDS
        const uint4* s1 = (const uint4*)w1p;
        uint4* d1 = (uint4*)lds;
        for (int i = tid; i < W1U / 4; i += 256) d1[i] = s1[i];
        const uint4* s2 = (const uint4*)w2p;
        uint4* d2 = (uint4*)(lds + W1U);
        for (int i = tid; i < W2U / 4; i += 256) d2[i] = s2[i];
    }
    __syncthreads();

    const long long row = (long long)blockIdx.x * 128 + wave * 16 + ln;
    const char* p0 = (const char*)selfp + row * (SBF ? 256 : 512);
    const char* p1;
    const char* p2;
    if (idx) {
        const int2 ii = *(const int2*)(idx + 2 * row);
        p1 = (const char*)other + (long long)ii.x * (OBF ? 256 : 512);
        p2 = (const char*)other + (long long)ii.y * (OBF ? 256 : 512);
    } else {
        p1 = (const char*)other + row * (OBF ? 256 : 512);
        p2 = p1;
    }

    // hidden^T accumulators, bias-initialized: c1[m][v] = b1[m*16 + 8*hf + v]
    v8f c1[8];
#pragma unroll
    for (int m = 0; m < 8; ++m)
        c1[m] = *(const v8f*)(b1 + m * 16 + hf * 8);

    // ---- layer 1: hidden^T = W1^T x X^T ----
#pragma unroll
    for (int kc = 0; kc < KC1; ++kc) {
        FragAB xb;
        const int off = (kc & 3) * 32 + hf * 16;   // B frag: K = kc*32+16h+e
        if (kc < 4)      load_bfrag<SBF>(xb, p0, off);
        else if (kc < 8) load_bfrag<OBF>(xb, p1, off);
        else             load_bfrag<OBF>(xb, p2, off);
        FragAB wa[8];
#pragma unroll
        for (int m = 0; m < 8; ++m)
            load_wfrag(wa[m], lds, ((kc * 8 + m) * 32 + lane) * 8);
#pragma unroll
        for (int m = 0; m < 8; ++m)
            c1[m] = __builtin_amdgcn_wmma_f32_16x16x32_bf16(
                false, wa[m].v, false, xb.v, (short)0, c1[m], false, false);
    }

    // ---- tanh, bf16-pack, half-wave exchange (C-layout -> B-frag) ----
    unsigned int pm[8][4], pp[8][4];
#pragma unroll
    for (int m = 0; m < 8; ++m) {
        v8f th;
#pragma unroll
        for (int v = 0; v < 8; ++v) th[v] = fast_tanh(c1[m][v]);
        U8bf u;
        u.v = __builtin_convertvector(th, v8bf);
#pragma unroll
        for (int j = 0; j < 4; ++j) pm[m][j] = u.w[j];
    }
#pragma unroll
    for (int m = 0; m < 8; ++m)
#pragma unroll
        for (int j = 0; j < 4; ++j)
            pp[m][j] = (unsigned int)__shfl_xor((int)pm[m][j], 16, 32);

    // out^T accumulators, bias-initialized
    v8f c2[8];
#pragma unroll
    for (int m = 0; m < 8; ++m)
        c2[m] = *(const v8f*)(b2 + m * 16 + hf * 8);

    // ---- layer 2: out^T = W2^T x hidden^T ----
#pragma unroll
    for (int kc = 0; kc < 4; ++kc) {
        FragAB hb;
#pragma unroll
        for (int j = 0; j < 4; ++j) {
            hb.w[j]     = hf ? pp[2 * kc + 1][j] : pm[2 * kc][j];
            hb.w[4 + j] = hf ? pm[2 * kc + 1][j] : pp[2 * kc][j];
        }
        FragAB wa[8];
#pragma unroll
        for (int m = 0; m < 8; ++m)
            load_wfrag(wa[m], lds, W1U + ((kc * 8 + m) * 32 + lane) * 8);
#pragma unroll
        for (int m = 0; m < 8; ++m)
            c2[m] = __builtin_amdgcn_wmma_f32_16x16x32_bf16(
                false, wa[m].v, false, hb.v, (short)0, c2[m], false, false);
    }

    // ---- store: lane writes row `row`, cols m*16 + 8*hf + v (contiguous) ----
    if constexpr (OUTBF) {
        unsigned short* ob = (unsigned short*)out + row * UDIM + hf * 8;
#pragma unroll
        for (int m = 0; m < 8; ++m) {
            U8bf u;
            u.v = __builtin_convertvector(c2[m], v8bf);
            *(uint4*)(ob + m * 16) = u.q;
        }
    } else {
        float* orow = (float*)out + row * UDIM + hf * 8;
#pragma unroll
        for (int m = 0; m < 8; ++m) {
            *(v8f*)(orow + m * 16) = c2[m];
        }
    }
}

// ---------------------------------------------------------------------------
extern "C" void kernel_launch(void* const* d_in, const int* in_sizes, int n_in,
                              void* d_out, int out_size, void* d_ws,
                              size_t ws_size, hipStream_t stream) {
    const float* h1 = (const float*)d_in[0];
    const float* h2 = (const float*)d_in[1];
    const float* h3 = (const float*)d_in[2];
    const float* h4 = (const float*)d_in[3];
    const int* idx2 = (const int*)d_in[4];
    const int* idx3 = (const int*)d_in[5];
    const int* idx4 = (const int*)d_in[6];
    const float* upW1[3] = {(const float*)d_in[7],  (const float*)d_in[11], (const float*)d_in[15]};
    const float* upb1[3] = {(const float*)d_in[8],  (const float*)d_in[12], (const float*)d_in[16]};
    const float* upW2[3] = {(const float*)d_in[9],  (const float*)d_in[13], (const float*)d_in[17]};
    const float* upb2[3] = {(const float*)d_in[10], (const float*)d_in[14], (const float*)d_in[18]};
    const float* dnW1[3] = {(const float*)d_in[19], (const float*)d_in[23], (const float*)d_in[27]};
    const float* dnb1[3] = {(const float*)d_in[20], (const float*)d_in[24], (const float*)d_in[28]};
    const float* dnW2[3] = {(const float*)d_in[21], (const float*)d_in[25], (const float*)d_in[29]};
    const float* dnb2[3] = {(const float*)d_in[22], (const float*)d_in[26], (const float*)d_in[30]};

    char* ws = (char*)d_ws;
    size_t off = 0;
    auto take = [&](size_t bytes) -> void* {
        off = (off + 1023) & ~(size_t)1023;
        void* p = ws + off;
        off += bytes;
        return p;
    };
    unsigned short *upW1p[3], *upW2p[3], *dnW1p[3], *dnW2p[3];
    for (int s = 0; s < 3; ++s) {
        upW1p[s] = (unsigned short*)take(49152 * 2);
        upW2p[s] = (unsigned short*)take(16384 * 2);
    }
    for (int s = 0; s < 3; ++s) {
        dnW1p[s] = (unsigned short*)take(32768 * 2);
        dnW2p[s] = (unsigned short*)take(16384 * 2);
    }
    // bf16 intermediates
    unsigned short* bufA = (unsigned short*)take((size_t)NN2 * UDIM * 2);  // h2n
    unsigned short* bufB = (unsigned short*)take((size_t)NN3 * UDIM * 2);  // h3n -> h2d
    unsigned short* bufC = (unsigned short*)take((size_t)NN4 * UDIM * 2);  // h4n -> h3d
    // f32 scatter accumulator (hd3 -> hd2 -> hd1)
    float* bufD = (float*)take((size_t)NN3 * UDIM * 4);

    for (int s = 0; s < 3; ++s) {
        pack_wT_kernel<<<12 * 16, 256, 0, stream>>>(upW1[s], upW1p[s], 12);
        pack_wT_kernel<<<4 * 16, 256, 0, stream>>>(upW2[s], upW2p[s], 4);
        pack_wT_kernel<<<8 * 16, 256, 0, stream>>>(dnW1[s], dnW1p[s], 8);
        pack_wT_kernel<<<4 * 16, 256, 0, stream>>>(dnW2[s], dnW2p[s], 4);
    }

    const size_t sh12 = (size_t)(12 * 8 * 32 * 8 + 4 * 8 * 32 * 8) * 4;  // 128 KB
    const size_t sh8  = (size_t)(8 * 8 * 32 * 8 + 4 * 8 * 32 * 8) * 4;   //  96 KB

    // ---- upward pass ----
    mlp_stage_kernel<12, false, false, true>
        <<<(unsigned)(NN2 / 128), 256, sh12, stream>>>(
            h2, h1, idx2, upb1[0], upb2[0], (const unsigned int*)upW1p[0],
            (const unsigned int*)upW2p[0], bufA);
    mlp_stage_kernel<12, false, true, true>
        <<<(unsigned)(NN3 / 128), 256, sh12, stream>>>(
            h3, bufA, idx3, upb1[1], upb2[1], (const unsigned int*)upW1p[1],
            (const unsigned int*)upW2p[1], bufB);
    mlp_stage_kernel<12, false, true, true>
        <<<(unsigned)(NN4 / 128), 256, sh12, stream>>>(
            h4, bufB, idx4, upb1[2], upb2[2], (const unsigned int*)upW1p[2],
            (const unsigned int*)upW2p[2], bufC);

    // ---- downward: level 3 ----
    zero4_kernel<<<(unsigned)(NN3 * UDIM / 4 / 256), 256, 0, stream>>>(
        (float4*)bufD, NN3 * UDIM / 4);
    scatter_add_bf16_kernel<<<(unsigned)(NN4 / 4), 256, 0, stream>>>(
        (const unsigned int*)bufC, idx4, bufD, NN4);
    mlp_stage_kernel<8, true, false, true>
        <<<(unsigned)(NN3 / 128), 256, sh8, stream>>>(
            bufB, bufD, nullptr, dnb1[2], dnb2[2], (const unsigned int*)dnW1p[2],
            (const unsigned int*)dnW2p[2], bufC);

    // ---- downward: level 2 ----
    zero4_kernel<<<(unsigned)(NN2 * UDIM / 4 / 256), 256, 0, stream>>>(
        (float4*)bufD, NN2 * UDIM / 4);
    scatter_add_bf16_kernel<<<(unsigned)(NN3 / 4), 256, 0, stream>>>(
        (const unsigned int*)bufC, idx3, bufD, NN3);
    mlp_stage_kernel<8, true, false, true>
        <<<(unsigned)(NN2 / 128), 256, sh8, stream>>>(
            bufA, bufD, nullptr, dnb1[1], dnb2[1], (const unsigned int*)dnW1p[1],
            (const unsigned int*)dnW2p[1], bufB);

    // ---- downward: level 1 -> d_out (f32) ----
    zero4_kernel<<<(unsigned)(NN1 * UDIM / 4 / 256), 256, 0, stream>>>(
        (float4*)bufD, NN1 * UDIM / 4);
    scatter_add_bf16_kernel<<<(unsigned)(NN2 / 4), 256, 0, stream>>>(
        (const unsigned int*)bufB, idx2, bufD, NN2);
    mlp_stage_kernel<8, false, false, false>
        <<<(unsigned)(NN1 / 128), 256, sh8, stream>>>(
            h1, bufD, nullptr, dnb1[0], dnb2[0], (const unsigned int*)dnW1p[0],
            (const unsigned int*)dnW2p[0], (float*)d_out);
}
